// Novel_Node_GAT_40346922779023
// MI455X (gfx1250) — matrix operand order repaired and checked by reading, not compiled, not run on hardware
//
#include <hip/hip_runtime.h>
#include <hip/hip_bf16.h>

typedef __attribute__((ext_vector_type(16))) _Float16 v16h;
typedef __attribute__((ext_vector_type(8)))  _Float16 v8h;
typedef __attribute__((ext_vector_type(8)))  float    v8f;

#define GAT_N   20000
#define GAT_E   320000
#define GAT_EF  (GAT_E + GAT_N)   // edges + self loops
#define GAT_D   128               // DIN == DH == 128
#define GAT_C   40
#define GAT_CP  48                // C padded to multiple of 16

// ---------------------------------------------------------------------------
// float <-> monotonically ordered u32 key (for atomicMax-based segment_max)
// ---------------------------------------------------------------------------
__device__ __forceinline__ unsigned fkey(float f) {
    unsigned b = __float_as_uint(f);
    return b ^ ((unsigned)((int)b >> 31) | 0x80000000u);
}
__device__ __forceinline__ float funkey(unsigned u) {
    unsigned b = (u & 0x80000000u) ? (u ^ 0x80000000u) : ~u;
    return __uint_as_float(b);
}

// ---------------------------------------------------------------------------
// generic u32 fill (zero-init sums, fill omax with key(-inf) = 0x007FFFFF)
// ---------------------------------------------------------------------------
__global__ void fill_u32_kernel(unsigned* __restrict__ p, unsigned val, int n) {
    int i = blockIdx.x * blockDim.x + threadIdx.x;
    if (i < n) p[i] = val;
}

// ---------------------------------------------------------------------------
// fp32 -> fp16 convert (x -> Xh for the first GEMM)
// ---------------------------------------------------------------------------
__global__ void f32_to_f16_kernel(const float* __restrict__ in,
                                  _Float16* __restrict__ out, int n) {
    int i = blockIdx.x * blockDim.x + threadIdx.x;
    if (i < n) out[i] = (_Float16)in[i];
}

// ---------------------------------------------------------------------------
// Pre-pack row-major fp32 W[K,Nw] into WMMA B-fragment order (f16):
//   Bp[((ntile*KC + kc)*32 + lane)*16 + j] = W[kc*32 + (lane/16)*16 + j][ntile*16 + lane%16]
// Zero-pads columns n >= Nw (used for the 40->48 padded output head).
// ---------------------------------------------------------------------------
__global__ void pack_w_kernel(const float* __restrict__ W,
                              _Float16* __restrict__ Bp,
                              int K, int Nw, int Npad) {
    int KC = K >> 5;
    int total = (Npad >> 4) * KC * 32 * 16;
    int idx = blockIdx.x * blockDim.x + threadIdx.x;
    if (idx >= total) return;
    int j     =  idx        & 15;
    int lane  = (idx >> 4)  & 31;
    int kc    = (idx >> 9) % KC;
    int ntile = (idx >> 9) / KC;
    int n = ntile * 16 + (lane & 15);
    int k = kc * 32 + (lane >> 4) * 16 + j;
    float v = (n < Nw) ? W[(size_t)k * Nw + n] : 0.0f;
    Bp[idx] = (_Float16)v;
}

// ---------------------------------------------------------------------------
// WMMA GEMM: C[M,ldc] = A[M,K] (f16 row-major) x Bp (packed f16), f32 accum.
// One wave per 16x16 output tile; K consumed in chunks of 32 via
// v_wmma_f32_16x16x32_f16. EXEC is all-ones (no divergence, 32-thread blocks,
// exact tiling: M=20000 is a multiple of 16).
// ---------------------------------------------------------------------------
__global__ __launch_bounds__(32)
void gemm_wmma_kernel(const _Float16* __restrict__ A,
                      const _Float16* __restrict__ Bp,
                      float* __restrict__ C,
                      int K, int ldc) {
    const int lane   = threadIdx.x & 31;
    const int half16 = lane >> 4;          // 0: lanes 0-15, 1: lanes 16-31
    const int mtile  = blockIdx.x;
    const int ntile  = blockIdx.y;
    const int KC     = K >> 5;

    const _Float16* arow = A + (size_t)(mtile * 16 + (lane & 15)) * K;
    const _Float16* bptr = Bp + (size_t)ntile * KC * 32 * 16 + (size_t)lane * 16;

    v8f c = {};
    for (int kc = 0; kc < KC; ++kc) {
        const int kb = kc * 32;
        // A fragment: lane holds row M=lane%16; two contiguous 8-half chunks.
        union { v16h v; v8h h[2]; } a;
        a.h[0] = *(const v8h*)(arow + kb + half16 * 16);
        a.h[1] = *(const v8h*)(arow + kb + half16 * 16 + 8);
        // B fragment: pre-packed contiguous 16 halves per lane.
        v16h b = *(const v16h*)(bptr + (size_t)kc * 32 * 16);
        c = __builtin_amdgcn_wmma_f32_16x16x32_f16(
                /*neg_a=*/false, a.v, /*neg_b=*/false, b,
                /*c_mod=*/(short)0, c, /*reuse_a=*/false, /*reuse_b=*/false);
    }
    // C/D layout: VGPR r -> row M = r + half16*8, col N = lane%16
    float* crow = C + (size_t)(mtile * 16 + half16 * 8) * ldc + ntile * 16 + (lane & 15);
    #pragma unroll
    for (int r = 0; r < 8; ++r) crow[(size_t)r * ldc] = c[r];
}

// ---------------------------------------------------------------------------
// Per-node attention dots: a_s[i] = h[i] . aW[0:128], a_t[i] = h[i] . aW[128:256]
// One wave per node, wave32 shuffle reduction.
// ---------------------------------------------------------------------------
__global__ __launch_bounds__(256)
void node_dots_kernel(const float* __restrict__ H, const float* __restrict__ aW,
                      float* __restrict__ a_s, float* __restrict__ a_t, int nnodes) {
    int lane = threadIdx.x & 31;
    int warp = threadIdx.x >> 5;
    int node = blockIdx.x * 8 + warp;
    if (node >= nnodes) return;
    const float* hr = H + (size_t)node * GAT_D;
    float ds = 0.f, dt = 0.f;
    #pragma unroll
    for (int i = lane; i < GAT_D; i += 32) {
        float hv = hr[i];
        ds += hv * aW[i];
        dt += hv * aW[GAT_D + i];
    }
    #pragma unroll
    for (int off = 16; off > 0; off >>= 1) {
        ds += __shfl_xor(ds, off, 32);
        dt += __shfl_xor(dt, off, 32);
    }
    if (lane == 0) { a_s[node] = ds; a_t[node] = dt; }
}

// ---------------------------------------------------------------------------
// Per-edge attention logits + segment sums (softmax denom over dst, sim sum
// over src). eid >= E are self-loops (src = dst = eid - E).
// ---------------------------------------------------------------------------
__global__ void edge_logits_kernel(const int* __restrict__ ei,
                                   const float* __restrict__ norm,
                                   const float* __restrict__ a_s,
                                   const float* __restrict__ a_t,
                                   const float* __restrict__ ab,
                                   float* __restrict__ e_buf,
                                   float* __restrict__ e_sum,
                                   float* __restrict__ s_sum) {
    int eid = blockIdx.x * blockDim.x + threadIdx.x;
    if (eid >= GAT_EF) return;
    int src = (eid < GAT_E) ? ei[eid]         : (eid - GAT_E);
    int dst = (eid < GAT_E) ? ei[GAT_E + eid] : (eid - GAT_E);
    float v = a_s[src] + a_t[dst] + ab[0];
    v = (v > 0.f) ? v : 0.2f * v;              // leaky_relu(0.2)
    float el = __expf(v);
    e_buf[eid] = el;
    atomicAdd(&e_sum[dst], el);
    atomicAdd(&s_sum[src], __expf(norm[eid]));
}

// ---------------------------------------------------------------------------
// Blend softmax attention with Jaccard-sim weight:
//   w = clip(alpha) * e/e_sum[dst] + (1-clip(alpha)) * exp(norm)/s_sum[dst]
// ---------------------------------------------------------------------------
__global__ void edge_weight_kernel(const int* __restrict__ ei,
                                   const float* __restrict__ norm,
                                   const float* __restrict__ e_buf,
                                   const float* __restrict__ e_sum,
                                   const float* __restrict__ s_sum,
                                   const float* __restrict__ alpha,
                                   float* __restrict__ w_buf) {
    int eid = blockIdx.x * blockDim.x + threadIdx.x;
    if (eid >= GAT_EF) return;
    int dst = (eid < GAT_E) ? ei[GAT_E + eid] : (eid - GAT_E);
    float al = fminf(fmaxf(alpha[0], 1e-4f), 0.9999f);
    float a  = e_buf[eid] / e_sum[dst];
    float sw = __expf(norm[eid]) / s_sum[dst];
    w_buf[eid] = al * a + (1.f - al) * sw;
}

// ---------------------------------------------------------------------------
// segment_max(w * h[src], dst) via ordered-u32 atomicMax, per (edge, feature).
// ---------------------------------------------------------------------------
__global__ void scatter_max_kernel(const int* __restrict__ ei,
                                   const float* __restrict__ H,
                                   const float* __restrict__ w_buf,
                                   unsigned* __restrict__ omax) {
    int idx = blockIdx.x * blockDim.x + threadIdx.x;
    if (idx >= GAT_EF * GAT_D) return;
    int f   = idx & (GAT_D - 1);
    int eid = idx >> 7;
    int src = (eid < GAT_E) ? ei[eid]         : (eid - GAT_E);
    int dst = (eid < GAT_E) ? ei[GAT_E + eid] : (eid - GAT_E);
    float val = w_buf[eid] * H[(size_t)src * GAT_D + f];
    atomicMax(&omax[(size_t)dst * GAT_D + f], fkey(val));
}

// ---------------------------------------------------------------------------
// Layer epilogue: decode max, + bias, ReLU, write f16 activations (input to
// the next WMMA GEMM).
// ---------------------------------------------------------------------------
__global__ void finalize_layer_kernel(const unsigned* __restrict__ omax,
                                      const float* __restrict__ bias,
                                      _Float16* __restrict__ Xh) {
    int idx = blockIdx.x * blockDim.x + threadIdx.x;
    if (idx >= GAT_N * GAT_D) return;
    float v = funkey(omax[idx]) + bias[idx & (GAT_D - 1)];
    v = fmaxf(v, 0.f);
    Xh[idx] = (_Float16)v;
}

// ---------------------------------------------------------------------------
// Output head epilogue: logits[:, :40] + bout -> log_softmax -> d_out.
// ---------------------------------------------------------------------------
__global__ void logsoftmax_kernel(const float* __restrict__ logits,
                                  const float* __restrict__ bout,
                                  float* __restrict__ out) {
    int node = blockIdx.x * blockDim.x + threadIdx.x;
    if (node >= GAT_N) return;
    const float* lr = logits + (size_t)node * GAT_CP;
    float v[GAT_C];
    float m = -1e30f;
    #pragma unroll
    for (int c = 0; c < GAT_C; ++c) { v[c] = lr[c] + bout[c]; m = fmaxf(m, v[c]); }
    float s = 0.f;
    #pragma unroll
    for (int c = 0; c < GAT_C; ++c) s += __expf(v[c] - m);
    float ls = __logf(s) + m;
    float* orow = out + (size_t)node * GAT_C;
    #pragma unroll
    for (int c = 0; c < GAT_C; ++c) orow[c] = v[c] - ls;
}

// ---------------------------------------------------------------------------

static inline int cdiv(long long a, int b) { return (int)((a + b - 1) / b); }

extern "C" void kernel_launch(void* const* d_in, const int* in_sizes, int n_in,
                              void* d_out, int out_size, void* d_ws, size_t ws_size,
                              hipStream_t stream) {
    const float* x      = (const float*)d_in[0];
    const int*   ei     = (const int*)  d_in[1];
    const float* norm   = (const float*)d_in[2];
    const float* W1     = (const float*)d_in[3];
    const float* aW1    = (const float*)d_in[4];
    const float* ab1    = (const float*)d_in[5];
    const float* alpha1 = (const float*)d_in[6];
    const float* b1     = (const float*)d_in[7];
    const float* W2     = (const float*)d_in[8];
    const float* aW2    = (const float*)d_in[9];
    const float* ab2    = (const float*)d_in[10];
    const float* alpha2 = (const float*)d_in[11];
    const float* b2     = (const float*)d_in[12];
    const float* Wout   = (const float*)d_in[13];
    const float* bout   = (const float*)d_in[14];
    float* out = (float*)d_out;

    // ---- workspace carve-up (aligned to 256B) ----
    char* ws = (char*)d_ws;
    size_t off = 0;
    auto carve = [&](size_t bytes) -> void* {
        void* p = ws + off;
        off = (off + bytes + 255) & ~(size_t)255;
        return p;
    };
    _Float16* Xh     = (_Float16*)carve((size_t)GAT_N * GAT_D * 2);  // activations (f16)
    float*    H      = (float*)   carve((size_t)GAT_N * GAT_D * 4);  // GEMM out (pre-aggr)
    _Float16* Wp1    = (_Float16*)carve(8 * 4 * 32 * 16 * 2);
    _Float16* Wp2    = (_Float16*)carve(8 * 4 * 32 * 16 * 2);
    _Float16* Wpo    = (_Float16*)carve(3 * 4 * 32 * 16 * 2);
    float*    a_s    = (float*)   carve((size_t)GAT_N * 4);
    float*    a_t    = (float*)   carve((size_t)GAT_N * 4);
    float*    e_buf  = (float*)   carve((size_t)GAT_EF * 4);
    float*    w_buf  = (float*)   carve((size_t)GAT_EF * 4);
    float*    e_sum  = (float*)   carve((size_t)GAT_N * 4);
    float*    s_sum  = (float*)   carve((size_t)GAT_N * 4);
    unsigned* omax   = (unsigned*)carve((size_t)GAT_N * GAT_D * 4);
    float*    logits = (float*)   carve((size_t)GAT_N * GAT_CP * 4);

    const int MT = GAT_N / 16;          // 1250 M tiles
    const int NT = GAT_D / 16;          // 8 N tiles

    // x -> f16 activations
    f32_to_f16_kernel<<<cdiv((long long)GAT_N * GAT_D, 256), 256, 0, stream>>>(
        x, Xh, GAT_N * GAT_D);

    auto run_layer = [&](const float* W, _Float16* Wp, const float* aW,
                         const float* ab, const float* alpha, const float* bias) {
        pack_w_kernel<<<cdiv(8 * 4 * 32 * 16, 256), 256, 0, stream>>>(
            W, Wp, GAT_D, GAT_D, GAT_D);
        gemm_wmma_kernel<<<dim3(MT, NT), 32, 0, stream>>>(Xh, Wp, H, GAT_D, GAT_D);
        node_dots_kernel<<<cdiv(GAT_N, 8), 256, 0, stream>>>(H, aW, a_s, a_t, GAT_N);
        fill_u32_kernel<<<cdiv(GAT_N, 256), 256, 0, stream>>>((unsigned*)e_sum, 0u, GAT_N);
        fill_u32_kernel<<<cdiv(GAT_N, 256), 256, 0, stream>>>((unsigned*)s_sum, 0u, GAT_N);
        fill_u32_kernel<<<cdiv((long long)GAT_N * GAT_D, 256), 256, 0, stream>>>(
            omax, 0x007FFFFFu, GAT_N * GAT_D);   // key(-inf)
        edge_logits_kernel<<<cdiv(GAT_EF, 256), 256, 0, stream>>>(
            ei, norm, a_s, a_t, ab, e_buf, e_sum, s_sum);
        edge_weight_kernel<<<cdiv(GAT_EF, 256), 256, 0, stream>>>(
            ei, norm, e_buf, e_sum, s_sum, alpha, w_buf);
        scatter_max_kernel<<<cdiv((long long)GAT_EF * GAT_D, 256), 256, 0, stream>>>(
            ei, H, w_buf, omax);
        finalize_layer_kernel<<<cdiv((long long)GAT_N * GAT_D, 256), 256, 0, stream>>>(
            omax, bias, Xh);                      // Xh <- relu(max + bias), feeds next GEMM
    };

    run_layer(W1, Wp1, aW1, ab1, alpha1, b1);
    run_layer(W2, Wp2, aW2, ab2, alpha2, b2);

    // output head: logits = h2 @ Wout (padded to 48 cols), then log_softmax
    pack_w_kernel<<<cdiv(3 * 4 * 32 * 16, 256), 256, 0, stream>>>(
        Wout, Wpo, GAT_D, GAT_C, GAT_CP);
    gemm_wmma_kernel<<<dim3(MT, GAT_CP / 16), 32, 0, stream>>>(Xh, Wpo, logits, GAT_D, GAT_CP);
    logsoftmax_kernel<<<cdiv(GAT_N, 128), 128, 0, stream>>>(logits, bout, out);

    (void)in_sizes; (void)n_in; (void)out_size; (void)ws_size;
}
